// A_24713241821238
// MI455X (gfx1250) — matrix operand order repaired and checked by reading, not compiled, not run on hardware
//
#include <hip/hip_runtime.h>

// Problem constants
#define BB   64
#define NN   197
#define DD   768
#define HH   12
#define HD   64
#define MM   (BB*NN)      // 12608 rows (= 394 * 32, exact)
#define NT13 13           // ceil(197/16)
#define KPAD 224          // 7*32, padded key dim for P@V
#define NSQ  (NN*NN)      // 38809

typedef __bf16 bf16_t;
typedef __attribute__((ext_vector_type(16))) __bf16 v16bf;
typedef __attribute__((ext_vector_type(8)))  __bf16 v8bf;
typedef __attribute__((ext_vector_type(8)))  float  v8f;

__device__ __forceinline__ v16bf cat8(v8bf lo, v8bf hi) {
    return __builtin_shufflevector(lo, hi, 0,1,2,3,4,5,6,7,8,9,10,11,12,13,14,15);
}
__device__ __forceinline__ v8f wmma_bf16(v16bf a, v16bf b, v8f c) {
    return __builtin_amdgcn_wmma_f32_16x16x32_bf16(false, a, false, b, (short)0, c, false, false);
}

// ---------------- fp32 -> bf16 convert ----------------
__global__ void k_cvt_bf16(const float* __restrict__ in, bf16_t* __restrict__ out, int n) {
    int i = blockIdx.x * blockDim.x + threadIdx.x;
    int stride = gridDim.x * blockDim.x;
    for (; i < n; i += stride) out[i] = (bf16_t)in[i];
}

// ---------------- Pre-resolve rel-pos bias: rpb[h][row][col] fp32 ----------------
__global__ void k_make_rpb(const float* __restrict__ rpt, const int* __restrict__ rpi,
                           float* __restrict__ rpb) {
    int i = blockIdx.x * blockDim.x + threadIdx.x;
    int stride = gridDim.x * blockDim.x;
    for (; i < HH * NSQ; i += stride) {
        const int h  = i / NSQ;
        const int rc = i % NSQ;
        rpb[i] = rpt[rpi[rc] * HH + h];
    }
}

// ---------------- QKV GEMM: [12608,768] x [768,2304] -> q/k/v bf16 [B,H,N,64] ----------------
// One wave computes a 32x64 fp32 tile (2 A row-tiles share 4 B fragments per k-step).
// qkv = x @ W^T, so the WMMA B-fragment column n (16 consecutive k) is a contiguous
// slice of W row n -- no transpose needed.
__global__ void k_qkv_gemm(const bf16_t* __restrict__ xb, const bf16_t* __restrict__ wb,
                           const float* __restrict__ qbias, const float* __restrict__ vbias,
                           bf16_t* __restrict__ qo, bf16_t* __restrict__ ko, bf16_t* __restrict__ vo) {
    const int lane = threadIdx.x & 31;
    const int half = lane >> 4;
    const int l16  = lane & 15;
    const int m0 = blockIdx.y * 32;
    const int n0 = blockIdx.x * 64;

    v8f acc[2][4] = {};
    const int rowA0 = m0 + l16;
    const int rowA1 = m0 + 16 + l16;
    for (int ks = 0; ks < 24; ++ks) {
        const int k0 = ks * 32;
        // A fragments (contiguous k order: lanes<16 -> K 0..15; lanes>=16 -> K 16..31)
        const bf16_t* ap0 = xb + (size_t)rowA0 * DD + k0 + half * 16;
        const bf16_t* ap1 = xb + (size_t)rowA1 * DD + k0 + half * 16;
        v16bf a0 = cat8(*(const v8bf*)ap0, *(const v8bf*)(ap0 + 8));
        v16bf a1 = cat8(*(const v8bf*)ap1, *(const v8bf*)(ap1 + 8));
#pragma unroll
        for (int nt = 0; nt < 4; ++nt) {
            const int col = n0 + nt * 16 + l16;
            // B fragment: lane column = col, 16 consecutive k (lanes>=16 hold K+16)
            const bf16_t* bp = wb + (size_t)col * DD + k0 + half * 16;
            v16bf b = cat8(*(const v8bf*)bp, *(const v8bf*)(bp + 8));
            acc[0][nt] = wmma_bf16(a0, b, acc[0][nt]);
            acc[1][nt] = wmma_bf16(a1, b, acc[1][nt]);
        }
    }
    // Epilogue: bias, q-scale, scatter to [B,H,N,64] bf16
#pragma unroll
    for (int nt = 0; nt < 4; ++nt) {
        const int col = n0 + nt * 16 + l16;
        const int s  = col / DD;          // 0=q 1=k 2=v
        const int cw = col % DD;
        const int h  = cw / HD, hd = cw % HD;
        const float badd = (s == 0) ? qbias[cw] : ((s == 2) ? vbias[cw] : 0.f);
        bf16_t* dst = (s == 0) ? qo : ((s == 1) ? ko : vo);
#pragma unroll
        for (int mt = 0; mt < 2; ++mt) {
#pragma unroll
            for (int r = 0; r < 8; ++r) {
                const int m = m0 + mt * 16 + r + half * 8;   // D layout row
                const int b = m / NN, n = m % NN;
                float v = acc[mt][nt][r] + badd;
                if (s == 0) v *= 0.125f;                     // 1/sqrt(64)
                dst[(((size_t)b * HH + h) * NN + n) * HD + hd] = (bf16_t)v;
            }
        }
    }
}

// ---------------- Fused attention per (b,h): QK^T + relpos + softmax + PV ----------------
__global__ void k_attn(const bf16_t* __restrict__ qb, const bf16_t* __restrict__ kb,
                       const bf16_t* __restrict__ vb,
                       const float* __restrict__ rpb,
                       bf16_t* __restrict__ ob) {
    extern __shared__ char smem_raw[];
    bf16_t* Ks = (bf16_t*)smem_raw;          // [KPAD][HD]   keys, zero-padded rows
    bf16_t* Vt = Ks + KPAD * HD;             // [HD][KPAD]   V transposed, zero-padded cols
    const int bh   = blockIdx.x;
    const int b    = bh / HH, h = bh % HH;
    const int tid  = threadIdx.x;
    const int lane = tid & 31;
    const int wave = tid >> 5;
    const int half = lane >> 4;
    const int l16  = lane & 15;
    bf16_t* Psw = Vt + HD * KPAD + (size_t)wave * 16 * KPAD;   // per-wave P tile [16][KPAD]

    const bf16_t* kbase = kb + (size_t)bh * NN * HD;
    const bf16_t* vbase = vb + (size_t)bh * NN * HD;
    for (int i = tid; i < KPAD * HD; i += blockDim.x) {
        const int kk = i / HD, d = i % HD;
        Ks[i] = (kk < NN) ? kbase[(size_t)kk * HD + d] : (bf16_t)0.f;
    }
    for (int i = tid; i < HD * KPAD; i += blockDim.x) {
        const int d = i / KPAD, kk = i % KPAD;
        Vt[i] = (kk < NN) ? vbase[(size_t)kk * HD + d] : (bf16_t)0.f;
    }
    __syncthreads();

    const bf16_t* qbase = qb + (size_t)bh * NN * HD;
    const float* rpbh = rpb + (size_t)h * NSQ;
    for (int qt = wave; qt < NT13; qt += 8) {
        const int q0 = qt * 16;
        const int rowA = min(q0 + l16, NN - 1);
        // Q fragments (K = 64 -> 2 steps of 32)
        v16bf aq[2];
#pragma unroll
        for (int ks = 0; ks < 2; ++ks) {
            const bf16_t* ap = qbase + (size_t)rowA * HD + ks * 32 + half * 16;
            aq[ks] = cat8(*(const v8bf*)ap, *(const v8bf*)(ap + 8));
        }
        // Scores S[16 x 208] in 13 accumulators
        v8f acc[NT13];
#pragma unroll
        for (int t = 0; t < NT13; ++t) {
            v8f c = {};
#pragma unroll
            for (int ks = 0; ks < 2; ++ks) {
                // B = K^T: column n = key index -> K row (contiguous in LDS)
                const bf16_t* bp = Ks + (size_t)(t * 16 + l16) * HD + ks * 32 + half * 16;
                c = wmma_bf16(aq[ks], cat8(*(const v8bf*)bp, *(const v8bf*)(bp + 8)), c);
            }
            acc[t] = c;
        }
        // Rel-pos bias + key masking (D layout: elem r -> row q0+r+8*half, col t*16+l16)
#pragma unroll
        for (int t = 0; t < NT13; ++t) {
            const int col = t * 16 + l16;
#pragma unroll
            for (int r = 0; r < 8; ++r) {
                const int row = q0 + r + half * 8;
                if (col < NN && row < NN) acc[t][r] += rpbh[row * NN + col];
                else                      acc[t][r] = -1e30f;
            }
        }
        // Row softmax: rows live in 16-lane halves -> shfl_xor masks 1,2,4,8
        float rmax[8], rsum[8];
#pragma unroll
        for (int r = 0; r < 8; ++r) {
            float m = -1e30f;
#pragma unroll
            for (int t = 0; t < NT13; ++t) m = fmaxf(m, acc[t][r]);
#pragma unroll
            for (int mk = 1; mk < 16; mk <<= 1) m = fmaxf(m, __shfl_xor(m, mk, 32));
            rmax[r] = m;
            rsum[r] = 0.f;
        }
#pragma unroll
        for (int t = 0; t < NT13; ++t)
#pragma unroll
            for (int r = 0; r < 8; ++r) {
                const float p = __expf(acc[t][r] - rmax[r]);
                acc[t][r] = p;
                rsum[r] += p;
            }
#pragma unroll
        for (int r = 0; r < 8; ++r) {
            float s = rsum[r];
#pragma unroll
            for (int mk = 1; mk < 16; mk <<= 1) s += __shfl_xor(s, mk, 32);
            rsum[r] = s;
        }
        // Spill P (bf16) to per-wave LDS in row-major for A-fragment reads
#pragma unroll
        for (int t = 0; t < NT13; ++t)
#pragma unroll
            for (int r = 0; r < 8; ++r)
                Psw[(r + half * 8) * KPAD + t * 16 + l16] = (bf16_t)acc[t][r];
        for (int i = lane; i < 256; i += 32)          // zero cols 208..223
            Psw[(i >> 4) * KPAD + 208 + (i & 15)] = (bf16_t)0.f;

        // O[16x64] = P[16x224] @ V[224x64]
        v8f o[4] = {};
#pragma unroll
        for (int ks = 0; ks < 7; ++ks) {
            const bf16_t* ap = Psw + (size_t)l16 * KPAD + ks * 32 + half * 16;
            v16bf a = cat8(*(const v8bf*)ap, *(const v8bf*)(ap + 8));
#pragma unroll
            for (int nt = 0; nt < 4; ++nt) {
                // B column n = head-dim index -> Vt row (contiguous)
                const bf16_t* bp = Vt + (size_t)(nt * 16 + l16) * KPAD + ks * 32 + half * 16;
                o[nt] = wmma_bf16(a, cat8(*(const v8bf*)bp, *(const v8bf*)(bp + 8)), o[nt]);
            }
        }
        // Normalize and store bf16 [B,N,D] with c = h*64 + hd
#pragma unroll
        for (int r = 0; r < 8; ++r) {
            const float inv = (rsum[r] > 0.f) ? 1.f / rsum[r] : 0.f;
            const int row = q0 + r + half * 8;
            if (row < NN) {
#pragma unroll
                for (int nt = 0; nt < 4; ++nt) {
                    const int col = h * HD + nt * 16 + l16;
                    ob[((size_t)b * NN + row) * DD + col] = (bf16_t)(o[nt][r] * inv);
                }
            }
        }
    }
}

// ---------------- Proj GEMM: [12608,768] x [768,768] + bias -> fp32 out ----------------
// One wave computes a 32x64 tile (2 A row-tiles share 4 B fragments).
__global__ void k_proj_gemm(const bf16_t* __restrict__ ob, const bf16_t* __restrict__ wb,
                            const float* __restrict__ pbias, float* __restrict__ out) {
    const int lane = threadIdx.x & 31;
    const int half = lane >> 4;
    const int l16  = lane & 15;
    const int m0 = blockIdx.y * 32;
    const int n0 = blockIdx.x * 64;

    v8f acc[2][4] = {};
    const int rowA0 = m0 + l16;
    const int rowA1 = m0 + 16 + l16;
    for (int ks = 0; ks < 24; ++ks) {
        const int k0 = ks * 32;
        const bf16_t* ap0 = ob + (size_t)rowA0 * DD + k0 + half * 16;
        const bf16_t* ap1 = ob + (size_t)rowA1 * DD + k0 + half * 16;
        v16bf a0 = cat8(*(const v8bf*)ap0, *(const v8bf*)(ap0 + 8));
        v16bf a1 = cat8(*(const v8bf*)ap1, *(const v8bf*)(ap1 + 8));
#pragma unroll
        for (int nt = 0; nt < 4; ++nt) {
            const int col = n0 + nt * 16 + l16;
            const bf16_t* bp = wb + (size_t)col * DD + k0 + half * 16;
            v16bf b = cat8(*(const v8bf*)bp, *(const v8bf*)(bp + 8));
            acc[0][nt] = wmma_bf16(a0, b, acc[0][nt]);
            acc[1][nt] = wmma_bf16(a1, b, acc[1][nt]);
        }
    }
#pragma unroll
    for (int nt = 0; nt < 4; ++nt) {
        const int col = n0 + nt * 16 + l16;
        const float bias = pbias[col];
#pragma unroll
        for (int mt = 0; mt < 2; ++mt) {
#pragma unroll
            for (int r = 0; r < 8; ++r) {
                const int m = m0 + mt * 16 + r + half * 8;
                out[(size_t)m * DD + col] = acc[mt][nt][r] + bias;
            }
        }
    }
}

extern "C" void kernel_launch(void* const* d_in, const int* in_sizes, int n_in,
                              void* d_out, int out_size, void* d_ws, size_t ws_size,
                              hipStream_t stream) {
    const float* x      = (const float*)d_in[0];
    const float* qkv_w  = (const float*)d_in[1];
    const float* q_bias = (const float*)d_in[2];
    const float* v_bias = (const float*)d_in[3];
    const float* rpt    = (const float*)d_in[4];
    const float* proj_w = (const float*)d_in[5];
    const float* proj_b = (const float*)d_in[6];
    const int*   rpi    = (const int*)d_in[7];
    float* out = (float*)d_out;

    // Workspace layout
    char* ws = (char*)d_ws;
    const size_t n_x   = (size_t)MM * DD;            // 9,682,944
    const size_t n_wq  = (size_t)3 * DD * DD;        // 1,769,472
    const size_t n_wp  = (size_t)DD * DD;            //   589,824
    const size_t n_hv  = (size_t)BB * HH * NN * HD;  // 9,682,944
    const size_t n_rpb = (size_t)HH * NSQ;           //   465,708
    bf16_t* xb   = (bf16_t*)ws;                 ws += n_x  * sizeof(bf16_t);
    bf16_t* wqb  = (bf16_t*)ws;                 ws += n_wq * sizeof(bf16_t);
    bf16_t* wpb  = (bf16_t*)ws;                 ws += n_wp * sizeof(bf16_t);
    bf16_t* qbuf = (bf16_t*)ws;                 ws += n_hv * sizeof(bf16_t);
    bf16_t* kbuf = (bf16_t*)ws;                 ws += n_hv * sizeof(bf16_t);
    bf16_t* vbuf = (bf16_t*)ws;                 ws += n_hv * sizeof(bf16_t);
    bf16_t* obuf = (bf16_t*)ws;                 ws += n_x  * sizeof(bf16_t);
    ws = (char*)(((uintptr_t)ws + 255) & ~(uintptr_t)255);
    float*  rpb  = (float*)ws;                  ws += n_rpb * sizeof(float);

    // 1) fp32 -> bf16 converts + rel-pos bias pre-resolve
    k_cvt_bf16<<<(int)((n_x  + 1023) / 1024), 1024, 0, stream>>>(x,      xb,  (int)n_x);
    k_cvt_bf16<<<(int)((n_wq + 1023) / 1024), 1024, 0, stream>>>(qkv_w,  wqb, (int)n_wq);
    k_cvt_bf16<<<(int)((n_wp + 1023) / 1024), 1024, 0, stream>>>(proj_w, wpb, (int)n_wp);
    k_make_rpb<<<(int)((n_rpb + 255) / 256), 256, 0, stream>>>(rpt, rpi, rpb);

    // 2) QKV GEMM (grid: 2304/64 x 12608/32), one wave per block, 32x64 tile
    k_qkv_gemm<<<dim3(36, MM / 32), 32, 0, stream>>>(xb, wqb, q_bias, v_bias, qbuf, kbuf, vbuf);

    // 3) Fused attention: one workgroup per (b,h); 8 waves; 112 KB dynamic LDS
    const size_t lds_bytes = (size_t)(KPAD * HD + HD * KPAD + 8 * 16 * KPAD) * sizeof(bf16_t);
    k_attn<<<BB * HH, 256, lds_bytes, stream>>>(qbuf, kbuf, vbuf, rpb, obuf);

    // 4) Proj GEMM + bias -> fp32 output, 32x64 tile per wave
    k_proj_gemm<<<dim3(12, MM / 32), 32, 0, stream>>>(obuf, wpb, proj_b, out);
}
